// HGDDTI_56899726737562
// MI455X (gfx1250) — compile-verified, matrix operand order checked by reading, not run on hardware
//
#include <hip/hip_runtime.h>
#include <hip/hip_bf16.h>

#define D_MODEL 256
#define NHEAD 8
#define HDIM 32

#define BM 64      // block tile rows
#define BN 128     // block tile cols
#define BK 32      // k step

typedef __bf16 v16bf __attribute__((ext_vector_type(16)));
typedef __bf16 v8bf  __attribute__((ext_vector_type(8)));
typedef float  v8f   __attribute__((ext_vector_type(8)));

__device__ __forceinline__ __bf16 f2bf(float f) {
  unsigned u = __builtin_bit_cast(unsigned, f);
  unsigned r = (u + 0x7FFFu + ((u >> 16) & 1u)) >> 16;  // round-to-nearest-even
  unsigned short s = (unsigned short)r;
  return __builtin_bit_cast(__bf16, s);
}

#define SHUF16(a, b) __builtin_shufflevector(a, b, 0,1,2,3,4,5,6,7,8,9,10,11,12,13,14,15)

// ---------------------------------------------------------------------------
// GEMM: C[M,N] = act(A[M,K] @ W[K,N] + bias). fp32 in/out, bf16 WMMA compute.
// Caller guarantees: M%64==0, N%128==0, K%32==0.
// Block: 256 threads = 8 waves, 64x128 macro-tile. Each wave owns a 2x2 grid
// of 16x16 WMMA tiles (4 accumulators) -> 4 v_wmma per 32-K step with A/B
// fragment reuse. A and W staged via LDS, converted fp32->bf16 once; B kept
// transposed [n][k] so every fragment is two contiguous 16B runs
// (ds_load_b128), matching ISA 7.12.2: lane l (half h) needs
// K = {8h..8h+7} u {16+8h..16+8h+7}.
// ---------------------------------------------------------------------------
__global__ void hg_wmma_gemm(const float* __restrict__ A,
                             const float* __restrict__ W,
                             const float* __restrict__ bias,
                             float* __restrict__ C,
                             int M, int N, int K, int act) {
  __shared__ __bf16 Asl[BM][BK];     // 4 KB
  __shared__ __bf16 BslT[BN][BK];    // 8 KB  [n][k]
  (void)M;

  int tid  = threadIdx.x;
  int lane = tid & 31;
  int wave = tid >> 5;      // 0..7
  int wmg  = wave >> 2;     // 0..1 -> 32-row group
  int wn   = wave & 3;      // 0..3 -> 16-col slot (and +64)
  int half = lane >> 4;     // 0/1
  int l    = lane & 15;

  int nbN = N / BN;
  int bm  = (blockIdx.x / nbN) * BM;
  int bn  = (blockIdx.x % nbN) * BN;

  // cooperative-fill indices
  int arow = tid >> 2;            // 0..63  (A row)
  int akq  = (tid & 3) * 8;       // 0,8,16,24 (8 k per thread)
  int bkr  = tid >> 3;            // 0..31  (B k-row)
  int bc0  = (tid & 7) * 16;      // 0,16,..112 (16 n per thread)

  const float* aptr = A + (size_t)(bm + arow) * K + akq;
  const float* bptr = W + (size_t)bkr * N + bn + bc0;

  v8f acc00 = {}, acc01 = {}, acc10 = {}, acc11 = {};

  for (int k0 = 0; k0 < K; k0 += BK) {
    // ---- stage A: 64x32 f32 -> bf16 (2x float4 load, one 16B ds store) ----
    float4 a0 = ((const float4*)(aptr + k0))[0];
    float4 a1 = ((const float4*)(aptr + k0))[1];
    __builtin_prefetch(aptr + k0 + BK);          // speculative; OOB dropped
    v8bf a8;
    a8[0] = f2bf(a0.x); a8[1] = f2bf(a0.y); a8[2] = f2bf(a0.z); a8[3] = f2bf(a0.w);
    a8[4] = f2bf(a1.x); a8[5] = f2bf(a1.y); a8[6] = f2bf(a1.z); a8[7] = f2bf(a1.w);
    *(v8bf*)&Asl[arow][akq] = a8;

    // ---- stage B: 32x128 f32, transposed bf16 stores ----
    const float* bp = bptr + (size_t)k0 * N;
    float4 b0 = ((const float4*)bp)[0];
    float4 b1 = ((const float4*)bp)[1];
    float4 b2 = ((const float4*)bp)[2];
    float4 b3 = ((const float4*)bp)[3];
    BslT[bc0 +  0][bkr] = f2bf(b0.x);
    BslT[bc0 +  1][bkr] = f2bf(b0.y);
    BslT[bc0 +  2][bkr] = f2bf(b0.z);
    BslT[bc0 +  3][bkr] = f2bf(b0.w);
    BslT[bc0 +  4][bkr] = f2bf(b1.x);
    BslT[bc0 +  5][bkr] = f2bf(b1.y);
    BslT[bc0 +  6][bkr] = f2bf(b1.z);
    BslT[bc0 +  7][bkr] = f2bf(b1.w);
    BslT[bc0 +  8][bkr] = f2bf(b2.x);
    BslT[bc0 +  9][bkr] = f2bf(b2.y);
    BslT[bc0 + 10][bkr] = f2bf(b2.z);
    BslT[bc0 + 11][bkr] = f2bf(b2.w);
    BslT[bc0 + 12][bkr] = f2bf(b3.x);
    BslT[bc0 + 13][bkr] = f2bf(b3.y);
    BslT[bc0 + 14][bkr] = f2bf(b3.z);
    BslT[bc0 + 15][bkr] = f2bf(b3.w);
    __syncthreads();

    // ---- fragments: two 16B runs each (ds_load_b128) ----
    int ra = wmg * 32 + l;
    v8bf x0 = *(const v8bf*)&Asl[ra     ][half * 8];
    v8bf x1 = *(const v8bf*)&Asl[ra     ][16 + half * 8];
    v8bf y0 = *(const v8bf*)&Asl[ra + 16][half * 8];
    v8bf y1 = *(const v8bf*)&Asl[ra + 16][16 + half * 8];
    int rb = wn * 16 + l;
    v8bf u0 = *(const v8bf*)&BslT[rb     ][half * 8];
    v8bf u1 = *(const v8bf*)&BslT[rb     ][16 + half * 8];
    v8bf t0 = *(const v8bf*)&BslT[rb + 64][half * 8];
    v8bf t1 = *(const v8bf*)&BslT[rb + 64][16 + half * 8];
    v16bf aF0 = SHUF16(x0, x1);
    v16bf aF1 = SHUF16(y0, y1);
    v16bf bF0 = SHUF16(u0, u1);
    v16bf bF1 = SHUF16(t0, t1);

    acc00 = __builtin_amdgcn_wmma_f32_16x16x32_bf16(false, aF0, false, bF0, (short)0, acc00, false, false);
    acc01 = __builtin_amdgcn_wmma_f32_16x16x32_bf16(false, aF0, false, bF1, (short)0, acc01, false, false);
    acc10 = __builtin_amdgcn_wmma_f32_16x16x32_bf16(false, aF1, false, bF0, (short)0, acc10, false, false);
    acc11 = __builtin_amdgcn_wmma_f32_16x16x32_bf16(false, aF1, false, bF1, (short)0, acc11, false, false);
    __syncthreads();
  }

  // ---- epilogue: C/D layout (lane -> n, vgpr r -> m, +8 for hi half) ----
  int cn0 = bn + wn * 16 + l;
  int cn1 = cn0 + 64;
  int cm0 = bm + wmg * 32 + half * 8;
  int cm1 = cm0 + 16;
  float bv0 = bias ? bias[cn0] : 0.0f;
  float bv1 = bias ? bias[cn1] : 0.0f;
#pragma unroll
  for (int r = 0; r < 8; ++r) {
    float v00 = acc00[r] + bv0;
    float v01 = acc01[r] + bv1;
    float v10 = acc10[r] + bv0;
    float v11 = acc11[r] + bv1;
    if (act == 1) {
      v00 = fmaxf(v00, 0.0f); v01 = fmaxf(v01, 0.0f);
      v10 = fmaxf(v10, 0.0f); v11 = fmaxf(v11, 0.0f);
    }
    C[(size_t)(cm0 + r) * N + cn0] = v00;
    C[(size_t)(cm0 + r) * N + cn1] = v01;
    C[(size_t)(cm1 + r) * N + cn0] = v10;
    C[(size_t)(cm1 + r) * N + cn1] = v11;
  }
}

// ---------------------------------------------------------------------------
// zero-pad x_d2d [N2D,44] -> [N2D,64]
// ---------------------------------------------------------------------------
__global__ void hg_pad44(const float* __restrict__ src, float* __restrict__ dst,
                         int rows) {
  int t = blockIdx.x * blockDim.x + threadIdx.x;
  if (t >= rows * 64) return;
  int r = t >> 6, c = t & 63;
  dst[t] = (c < 44) ? src[r * 44 + c] : 0.0f;
}

// ---------------------------------------------------------------------------
// GAT helper kernels
// ---------------------------------------------------------------------------
__global__ void hg_fill(float* __restrict__ p, float v, long long n) {
  long long t = (long long)blockIdx.x * blockDim.x + threadIdx.x;
  if (t < n) p[t] = v;
}

__global__ void hg_att_score(const float* __restrict__ H,
                             const float* __restrict__ a_src,
                             const float* __restrict__ a_dst,
                             float* __restrict__ as_, float* __restrict__ ad_,
                             int Nn) {
  int t = blockIdx.x * blockDim.x + threadIdx.x;
  if (t >= Nn * NHEAD) return;
  int n = t / NHEAD, h = t % NHEAD;
  const float* hp = H + (size_t)n * D_MODEL + h * HDIM;
  float s = 0.f, d = 0.f;
#pragma unroll
  for (int c = 0; c < HDIM; ++c) {
    float x = hp[c];
    s += x * a_src[h * HDIM + c];
    d += x * a_dst[h * HDIM + c];
  }
  as_[t] = s;
  ad_[t] = d;
}

__device__ __forceinline__ void edge_sd(const int* __restrict__ ei, int E,
                                        long long e, int& s, int& d) {
  if (e < E) { s = ei[e]; d = ei[E + e]; }
  else       { s = d = (int)(e - E); }          // implicit self loop
}

__device__ __forceinline__ void atomicMaxF(float* addr, float val) {
  unsigned* ua = (unsigned*)addr;
  unsigned old = __builtin_bit_cast(unsigned, *addr);
  while (__builtin_bit_cast(float, old) < val) {
    unsigned assumed = old;
    old = atomicCAS(ua, assumed, __builtin_bit_cast(unsigned, val));
    if (old == assumed) break;
  }
}

__global__ void hg_edge_max(const int* __restrict__ ei, int E, int Nn,
                            const float* __restrict__ as_,
                            const float* __restrict__ ad_,
                            float* __restrict__ m) {
  long long e = (long long)blockIdx.x * blockDim.x + threadIdx.x;
  if (e >= (long long)E + Nn) return;
  int s, d; edge_sd(ei, E, e, s, d);
#pragma unroll
  for (int h = 0; h < NHEAD; ++h) {
    float v = as_[s * NHEAD + h] + ad_[d * NHEAD + h];
    v = v > 0.f ? v : 0.2f * v;                 // leaky_relu(0.2)
    atomicMaxF(&m[d * NHEAD + h], v);
  }
}

__global__ void hg_edge_expsum(const int* __restrict__ ei, int E, int Nn,
                               const float* __restrict__ as_,
                               const float* __restrict__ ad_,
                               const float* __restrict__ m,
                               float* __restrict__ den) {
  long long e = (long long)blockIdx.x * blockDim.x + threadIdx.x;
  if (e >= (long long)E + Nn) return;
  int s, d; edge_sd(ei, E, e, s, d);
#pragma unroll
  for (int h = 0; h < NHEAD; ++h) {
    float v = as_[s * NHEAD + h] + ad_[d * NHEAD + h];
    v = v > 0.f ? v : 0.2f * v;
    atomicAdd(&den[d * NHEAD + h], expf(v - m[d * NHEAD + h]));
  }
}

__global__ void hg_edge_scatter(const int* __restrict__ ei, int E, int Nn,
                                const float* __restrict__ as_,
                                const float* __restrict__ ad_,
                                const float* __restrict__ m,
                                const float* __restrict__ den,
                                const float* __restrict__ H,
                                float* __restrict__ O) {
  long long gid = (long long)blockIdx.x * blockDim.x + threadIdx.x;
  long long tot = ((long long)E + Nn) * D_MODEL;
  if (gid >= tot) return;
  long long e = gid >> 8;
  int c = (int)(gid & 255);
  int h = c >> 5;
  int s, d; edge_sd(ei, E, e, s, d);
  float v = as_[s * NHEAD + h] + ad_[d * NHEAD + h];
  v = v > 0.f ? v : 0.2f * v;
  float ex = expf(v - m[d * NHEAD + h]);
  float alpha = ex / (den[d * NHEAD + h] + 1e-16f);
  atomicAdd(&O[(size_t)d * D_MODEL + c], H[(size_t)s * D_MODEL + c] * alpha);
}

__global__ void hg_bias_relu(float* __restrict__ p, const float* __restrict__ bias,
                             long long n) {
  long long t = (long long)blockIdx.x * blockDim.x + threadIdx.x;
  if (t >= n) return;
  p[t] = fmaxf(p[t] + bias[t & 255], 0.0f);
}

// ---------------------------------------------------------------------------
// Fusion helpers
// ---------------------------------------------------------------------------
__global__ void hg_gather_rows(const float* __restrict__ X,
                               const int* __restrict__ sup,
                               float* __restrict__ V) {
  int t = blockIdx.x * blockDim.x + threadIdx.x;
  if (t >= 128 * 256) return;
  int b = t >> 8, c = t & 255;
  V[t] = X[(size_t)sup[b] * D_MODEL + c];
}

__global__ void hg_fused_cat(const float* __restrict__ seq,
                             const float* __restrict__ pseq,
                             const float* __restrict__ v2,
                             const float* __restrict__ v3,
                             const float* __restrict__ vp,
                             float* __restrict__ fused) {
  int t = blockIdx.x * blockDim.x + threadIdx.x;
  if (t >= 128 * 1280) return;
  int b = t / 1280, c = t % 1280;
  float v;
  if (c < 256)       v = seq [b * 256 + c];
  else if (c < 512)  v = pseq[b * 256 + c - 256];
  else if (c < 768)  v = v2  [b * 256 + c - 512];
  else if (c < 1024) v = v3  [b * 256 + c - 768];
  else               v = vp  [b * 256 + c - 1024];
  fused[t] = v;
}

__global__ void hg_fus2(const float* __restrict__ F1, const float* __restrict__ W2,
                        const float* __restrict__ b2, float* __restrict__ out) {
  int b = blockIdx.x * blockDim.x + threadIdx.x;
  if (b >= 128) return;
  float s = 0.f;
#pragma unroll 8
  for (int c = 0; c < 256; ++c) s += F1[b * 256 + c] * W2[c];
  out[b] = s + b2[0];
}

// ---------------------------------------------------------------------------
static inline int cdiv(long long a, long long b) { return (int)((a + b - 1) / b); }

extern "C" void kernel_launch(void* const* d_in, const int* in_sizes, int n_in,
                              void* d_out, int out_size, void* d_ws, size_t ws_size,
                              hipStream_t stream) {
  (void)in_sizes; (void)n_in; (void)out_size; (void)ws_size;
  const int N2D = 5248, E2D = 12800;
  const int N3D = 3968, E3D = 9600;
  const int NP  = 51328, EP = 615936;

  int i = 0;
  const float* x_d2d       = (const float*)d_in[i++];   // 0
  const float* x_d3d       = (const float*)d_in[i++];   // 1
  const float* x_p         = (const float*)d_in[i++];   // 2
  const float* drug_seq    = (const float*)d_in[i++];   // 3
  const float* protein_esm = (const float*)d_in[i++];   // 4
  // params (pytree leaves in dict-insertion order)
  const float* atom_W  = (const float*)d_in[i++];  const float* atom_b  = (const float*)d_in[i++];
  const float* seq1_W  = (const float*)d_in[i++];  const float* seq1_b  = (const float*)d_in[i++];
  const float* seq2_W  = (const float*)d_in[i++];  const float* seq2_b  = (const float*)d_in[i++];
  const float* esm_W   = (const float*)d_in[i++];  const float* esm_b   = (const float*)d_in[i++];
  const float* d2c1_W  = (const float*)d_in[i++];  const float* d2c1_as = (const float*)d_in[i++];
  const float* d2c1_ad = (const float*)d_in[i++];  const float* d2c1_b  = (const float*)d_in[i++];
  const float* d2c2_W  = (const float*)d_in[i++];  const float* d2c2_as = (const float*)d_in[i++];
  const float* d2c2_ad = (const float*)d_in[i++];  const float* d2c2_b  = (const float*)d_in[i++];
  const float* d3_W    = (const float*)d_in[i++];  const float* d3_b    = (const float*)d_in[i++];
  const float* d3c1_W  = (const float*)d_in[i++];  const float* d3c1_as = (const float*)d_in[i++];
  const float* d3c1_ad = (const float*)d_in[i++];  const float* d3c1_b  = (const float*)d_in[i++];
  const float* d3c2_W  = (const float*)d_in[i++];  const float* d3c2_as = (const float*)d_in[i++];
  const float* d3c2_ad = (const float*)d_in[i++];  const float* d3c2_b  = (const float*)d_in[i++];
  const float* p_W     = (const float*)d_in[i++];  const float* p_b     = (const float*)d_in[i++];
  const float* pc1_W   = (const float*)d_in[i++];  const float* pc1_as  = (const float*)d_in[i++];
  const float* pc1_ad  = (const float*)d_in[i++];  const float* pc1_b   = (const float*)d_in[i++];
  const float* pc2_W   = (const float*)d_in[i++];  const float* pc2_as  = (const float*)d_in[i++];
  const float* pc2_ad  = (const float*)d_in[i++];  const float* pc2_b   = (const float*)d_in[i++];
  const float* fus1_W  = (const float*)d_in[i++];  const float* fus1_b  = (const float*)d_in[i++];
  const float* fus2_W  = (const float*)d_in[i++];  const float* fus2_b  = (const float*)d_in[i++];
  const int* ei_d2d = (const int*)d_in[i++];
  const int* ei_d3d = (const int*)d_in[i++];
  const int* ei_p   = (const int*)d_in[i++];
  const int* sup_2d = (const int*)d_in[i++];
  const int* sup_3d = (const int*)d_in[i++];
  const int* sup_p  = (const int*)d_in[i++];

  // workspace layout (floats); big region sized for protein, reused per branch
  float* w = (float*)d_ws;
  float* X    = w; w += (size_t)NP * D_MODEL;
  float* H    = w; w += (size_t)NP * D_MODEL;
  float* O    = w; w += (size_t)NP * D_MODEL;
  float* aS   = w; w += (size_t)NP * NHEAD;
  float* aD   = w; w += (size_t)NP * NHEAD;
  float* mM   = w; w += (size_t)NP * NHEAD;
  float* den  = w; w += (size_t)NP * NHEAD;
  float* x2p  = w; w += (size_t)N2D * 64;     // padded x_d2d (K 44 -> 64)
  float* seqt = w; w += 128 * 512;
  float* seqv = w; w += 128 * 256;
  float* pseq = w; w += 128 * 256;
  float* v2   = w; w += 128 * 256;
  float* v3   = w; w += 128 * 256;
  float* vp   = w; w += 128 * 256;
  float* fused= w; w += 128 * 1280;
  float* f1   = w; w += 128 * 256;

  auto gemm = [&](const float* A, const float* Wt, const float* bias, float* Cc,
                  int M, int N, int K, int act) {
    int blocks = (M / BM) * (N / BN);
    hg_wmma_gemm<<<blocks, 256, 0, stream>>>(A, Wt, bias, Cc, M, N, K, act);
  };

  auto gat = [&](const float* Xin, float* Hbuf, float* Obuf,
                 const int* ei, int E, int Nn,
                 const float* Wt, const float* asrc, const float* adst,
                 const float* bias) {
    gemm(Xin, Wt, nullptr, Hbuf, Nn, D_MODEL, D_MODEL, 0);
    hg_att_score<<<cdiv((long long)Nn * NHEAD, 256), 256, 0, stream>>>(
        Hbuf, asrc, adst, aS, aD, Nn);
    hg_fill<<<cdiv((long long)Nn * NHEAD, 256), 256, 0, stream>>>(mM, -1e30f, (long long)Nn * NHEAD);
    hg_fill<<<cdiv((long long)Nn * NHEAD, 256), 256, 0, stream>>>(den, 0.0f, (long long)Nn * NHEAD);
    hg_fill<<<cdiv((long long)Nn * D_MODEL, 256), 256, 0, stream>>>(Obuf, 0.0f, (long long)Nn * D_MODEL);
    long long Et = (long long)E + Nn;
    hg_edge_max   <<<cdiv(Et, 256), 256, 0, stream>>>(ei, E, Nn, aS, aD, mM);
    hg_edge_expsum<<<cdiv(Et, 256), 256, 0, stream>>>(ei, E, Nn, aS, aD, mM, den);
    hg_edge_scatter<<<cdiv(Et * D_MODEL, 256), 256, 0, stream>>>(
        ei, E, Nn, aS, aD, mM, den, Hbuf, Obuf);
    hg_bias_relu<<<cdiv((long long)Nn * D_MODEL, 256), 256, 0, stream>>>(
        Obuf, bias, (long long)Nn * D_MODEL);
  };

  // ---- drug fingerprint MLP + protein ESM projection ----
  gemm(drug_seq, seq1_W, seq1_b, seqt, 128, 512, 2048, 1);
  gemm(seqt, seq2_W, seq2_b, seqv, 128, 256, 512, 0);
  gemm(protein_esm, esm_W, esm_b, pseq, 128, 256, 1280, 1);

  // ---- drug 2D branch (pad K 44 -> 64 so GEMM needs no tail masking) ----
  hg_pad44<<<cdiv((long long)N2D * 64, 256), 256, 0, stream>>>(x_d2d, x2p, N2D);
  // atom_W is [44,256]; zero-extend it to [64,256] (reuse H as staging).
  hg_fill<<<cdiv(64 * 256, 256), 256, 0, stream>>>(H, 0.0f, 64 * 256);
  hipMemcpyAsync(H, atom_W, sizeof(float) * 44 * 256, hipMemcpyDeviceToDevice, stream);
  gemm(x2p, H, atom_b, X, N2D, 256, 64, 1);
  gat(X, H, O, ei_d2d, E2D, N2D, d2c1_W, d2c1_as, d2c1_ad, d2c1_b);
  gat(O, H, X, ei_d2d, E2D, N2D, d2c2_W, d2c2_as, d2c2_ad, d2c2_b);
  hg_gather_rows<<<cdiv(128 * 256, 256), 256, 0, stream>>>(X, sup_2d, v2);

  // ---- drug 3D branch ----
  gemm(x_d3d, d3_W, d3_b, X, N3D, 256, 64, 1);
  gat(X, H, O, ei_d3d, E3D, N3D, d3c1_W, d3c1_as, d3c1_ad, d3c1_b);
  gat(O, H, X, ei_d3d, E3D, N3D, d3c2_W, d3c2_as, d3c2_ad, d3c2_b);
  hg_gather_rows<<<cdiv(128 * 256, 256), 256, 0, stream>>>(X, sup_3d, v3);

  // ---- protein branch ----
  gemm(x_p, p_W, p_b, X, NP, 256, 1280, 1);
  gat(X, H, O, ei_p, EP, NP, pc1_W, pc1_as, pc1_ad, pc1_b);
  gat(O, H, X, ei_p, EP, NP, pc2_W, pc2_as, pc2_ad, pc2_b);
  hg_gather_rows<<<cdiv(128 * 256, 256), 256, 0, stream>>>(X, sup_p, vp);

  // ---- fusion ----
  hg_fused_cat<<<cdiv(128 * 1280, 256), 256, 0, stream>>>(seqv, pseq, v2, v3, vp, fused);
  gemm(fused, fus1_W, fus1_b, f1, 128, 256, 1280, 1);
  hg_fus2<<<1, 128, 0, stream>>>(f1, fus2_W, fus2_b, (float*)d_out);
}
